// Mutual_Attention2D_13013750907332
// MI455X (gfx1250) — matrix operand order repaired and checked by reading, not compiled, not run on hardware
//
#include <hip/hip_runtime.h>
#include <stdint.h>
#include <math.h>

// ---------------- types for WMMA ----------------
typedef __attribute__((ext_vector_type(16))) __bf16        v16bf;
typedef __attribute__((ext_vector_type(8)))  float         v8f;
typedef __attribute__((ext_vector_type(16))) unsigned short v16u;
typedef __attribute__((ext_vector_type(8)))  unsigned short u16x8;

// ---------------- problem constants ----------------
#define B_     8
#define C_     192
#define H_     128
#define W_     128
#define HW_    16384            // 128*128
#define HEADS_ 6
#define CH_    32               // C_/HEADS_
#define EPS_   1e-12f
#define KSPLIT_ 32              // K-split factor for q.k^T

static constexpr size_t NQKV = (size_t)B_ * C_ * HW_;       // elems per q/k/v buffer
static constexpr size_t NMB  = (size_t)B_ * C_ * C_;        // M matrices (bf16)
static constexpr int    NSSQ = B_ * C_;                     // per-row sumsq
static constexpr int    NS   = B_ * HEADS_ * CH_ * CH_;     // attn logits (f32)

// ---------------- helpers ----------------
__device__ __forceinline__ unsigned short f2bf(float f) {
    unsigned u = __float_as_uint(f);
    u += 0x7fffu + ((u >> 16) & 1u);     // round-to-nearest-even
    return (unsigned short)(u >> 16);
}

__device__ __forceinline__ v8f wmma_bf16(v16bf a, v16bf b, v8f c) {
    return __builtin_amdgcn_wmma_f32_16x16x32_bf16(
        false, a, false, b, (short)0, c, false, false);
}

// A-matrix 16x32 bf16 fragment, row-major source (row m0+lane%16, K base k0).
// Lanes 0-15 hold K 0..15, lanes 16-31 hold K 16..31 (same order as B fragments).
__device__ __forceinline__ v16bf fragA(const unsigned short* __restrict__ base,
                                       int ld, int m0, int k0, int lane) {
    int m    = m0 + (lane & 15);
    int half = (lane >> 4) * 16;
    const unsigned short* p = base + (size_t)m * ld + k0;
    u16x8 lo = *(const u16x8*)(p + half);
    u16x8 hi = *(const u16x8*)(p + 8 + half);
    v16u u;
#pragma unroll
    for (int i = 0; i < 8; ++i) { u[i] = lo[i]; u[8 + i] = hi[i]; }
    return __builtin_bit_cast(v16bf, u);
}

// B-matrix 32x16 bf16 fragment for D = A * B^T style use: B[k][n] = src[n][k]
// (row n0+lane%16 of src, 16 contiguous K starting at k0 + (lane/16)*16).
// Contiguous order used here (same as fragA): lanes 0-15 hold K 0..15, lanes 16-31 hold K 16..31, column N = lane%16.
__device__ __forceinline__ v16bf fragBt(const unsigned short* __restrict__ base,
                                        int ld, int n0, int k0, int lane) {
    int n  = n0 + (lane & 15);
    int kb = (lane >> 4) * 16;
    const unsigned short* p = base + (size_t)n * ld + k0 + kb;
    u16x8 lo = *(const u16x8*)(p);
    u16x8 hi = *(const u16x8*)(p + 8);
    v16u u;
#pragma unroll
    for (int i = 0; i < 8; ++i) { u[i] = lo[i]; u[8 + i] = hi[i]; }
    return __builtin_bit_cast(v16bf, u);
}

// B fragment from an LDS tile that is ALREADY transposed to [n][TLD] layout:
// per lane a contiguous 16-element K run -> two aligned ds_load_b128.
__device__ __forceinline__ v16bf fragBt_lds(const unsigned short* lds,
                                            int ldl, int n0, int lane) {
    int n  = n0 + (lane & 15);
    int kb = (lane >> 4) * 16;
    const unsigned short* p = lds + n * ldl + kb;
    u16x8 lo = *(const u16x8*)(p);
    u16x8 hi = *(const u16x8*)(p + 8);
    v16u u;
#pragma unroll
    for (int i = 0; i < 8; ++i) { u[i] = lo[i]; u[8 + i] = hi[i]; }
    return __builtin_bit_cast(v16bf, u);
}

// ---------------- kernel 0: zero small f32 workspace regions ----------------
__global__ void zero_kernel(float* __restrict__ p, int n) {
    int i = blockIdx.x * 256 + threadIdx.x;
    if (i < n) p[i] = 0.0f;
}

// ---------------- kernel 1: fused depthwise conv -> bf16 q/k/v + sumsq ------
// grid: B_ * 576 * 64 blocks of 256 threads; block = (b, cc, 256-pixel chunk)
__global__ __launch_bounds__(256) void dwconv_kernel(
    const float* __restrict__ x, const float* __restrict__ y,
    const float* __restrict__ wdw,
    unsigned short* __restrict__ Qb, unsigned short* __restrict__ Kb,
    unsigned short* __restrict__ Vb,
    float* __restrict__ sumsq_q, float* __restrict__ sumsq_k)
{
    int blk   = blockIdx.x;
    int chunk = blk & 63;
    int cc    = (blk >> 6) % (3 * C_);
    int b     = blk / (64 * 3 * C_);
    int part  = cc / C_;                 // 0:q<-x, 1:k<-y, 2:v<-y
    int c     = cc - part * C_;

    const float* src = (part == 0 ? x : y) + ((size_t)b * C_ + c) * HW_;
    const float* wt  = wdw + cc * 9;     // uniform per block -> scalar loads

    int p  = chunk * 256 + threadIdx.x;
    int hh = p >> 7, ww = p & 127;

    float acc = 0.0f;
#pragma unroll
    for (int di = -1; di <= 1; ++di) {
        int h2 = hh + di;
        if ((unsigned)h2 < (unsigned)H_) {
#pragma unroll
            for (int dj = -1; dj <= 1; ++dj) {
                int w2 = ww + dj;
                if ((unsigned)w2 < (unsigned)W_)
                    acc += src[h2 * W_ + w2] * wt[(di + 1) * 3 + (dj + 1)];
            }
        }
    }

    unsigned short* dst = (part == 0) ? Qb : (part == 1) ? Kb : Vb;
    dst[((size_t)b * C_ + c) * HW_ + p] = f2bf(acc);

    if (part < 2) {
        float ss = acc * acc;            // full-precision row sumsq for L2 norm
#pragma unroll
        for (int o = 16; o > 0; o >>= 1) ss += __shfl_xor(ss, o, 32);
        if ((threadIdx.x & 31) == 0) {
            float* tgt = (part == 0 ? sumsq_q : sumsq_k) + b * C_ + c;
            unsafeAtomicAdd(tgt, ss);
        }
    }
}

// ---------------- kernel 2: S = Q_raw . K_raw^T via WMMA, K-split ------------
// grid: 48*KSPLIT_ blocks of 32 threads (one full wave: EXEC all-ones for WMMA)
__global__ __launch_bounds__(32) void attn_qk_kernel(
    const unsigned short* __restrict__ Qb,
    const unsigned short* __restrict__ Kb,
    float* __restrict__ S)
{
    const int SLICE = HW_ / KSPLIT_;     // 512
    int blk  = blockIdx.x;
    int ks   = blk & (KSPLIT_ - 1);
    int bh   = blk >> 5;                 // 0..47  (b*6+h); rows bh*32..+31
    int lane = threadIdx.x & 31;

    const unsigned short* Qbase = Qb + (size_t)bh * CH_ * HW_;
    const unsigned short* Kbase = Kb + (size_t)bh * CH_ * HW_;

    v8f a00 = {}, a01 = {}, a10 = {}, a11 = {};
    int kend = ks * SLICE + SLICE;
    for (int k = ks * SLICE; k < kend; k += 32) {
        v16bf qa0 = fragA (Qbase, HW_, 0,  k, lane);
        v16bf qa1 = fragA (Qbase, HW_, 16, k, lane);
        v16bf kb0 = fragBt(Kbase, HW_, 0,  k, lane);
        v16bf kb1 = fragBt(Kbase, HW_, 16, k, lane);
        a00 = wmma_bf16(qa0, kb0, a00);
        a01 = wmma_bf16(qa0, kb1, a01);
        a10 = wmma_bf16(qa1, kb0, a10);
        a11 = wmma_bf16(qa1, kb1, a11);
    }

    float* Sbh = S + (size_t)bh * CH_ * CH_;
    int half = lane >> 4, n = lane & 15;
#pragma unroll
    for (int r = 0; r < 8; ++r) {
        int m = r + 8 * half;
        unsafeAtomicAdd(&Sbh[(m)      * CH_ + n],      a00[r]);
        unsafeAtomicAdd(&Sbh[(m)      * CH_ + 16 + n], a01[r]);
        unsafeAtomicAdd(&Sbh[(16 + m) * CH_ + n],      a10[r]);
        unsafeAtomicAdd(&Sbh[(16 + m) * CH_ + 16 + n], a11[r]);
    }
}

// ---------------- kernel 3: scale by norms & temperature, softmax ------------
// grid: 48 blocks of 32 threads; lane = row c
__global__ __launch_bounds__(32) void softmax_kernel(
    float* __restrict__ S,
    const float* __restrict__ sumsq_q, const float* __restrict__ sumsq_k,
    const float* __restrict__ temp)
{
    int bh = blockIdx.x;
    int b  = bh / HEADS_, h = bh - b * HEADS_;
    int c  = threadIdx.x;

    __shared__ float nks[CH_];
    nks[c] = fmaxf(sqrtf(sumsq_k[b * C_ + h * CH_ + c]), EPS_);
    __syncthreads();

    float rs = temp[h] / fmaxf(sqrtf(sumsq_q[b * C_ + h * CH_ + c]), EPS_);
    float* row = S + ((size_t)bh * CH_ + c) * CH_;

    float v[CH_];
    float mx = -3.4e38f;
#pragma unroll
    for (int d = 0; d < CH_; ++d) {
        v[d] = row[d] * rs / nks[d];
        mx = fmaxf(mx, v[d]);
    }
    float sum = 0.0f;
#pragma unroll
    for (int d = 0; d < CH_; ++d) { v[d] = __expf(v[d] - mx); sum += v[d]; }
    float inv = 1.0f / sum;
#pragma unroll
    for (int d = 0; d < CH_; ++d) row[d] = v[d] * inv;
}

// ---------------- kernel 4: M[b] = w_proj * blockdiag(attn) -> bf16 ----------
// grid: B_*C_ blocks of 192 threads; thread = output column cd
__global__ __launch_bounds__(192) void buildm_kernel(
    const float* __restrict__ wproj, const float* __restrict__ S,
    unsigned short* __restrict__ Mb)
{
    int b  = blockIdx.x / C_, o = blockIdx.x % C_;
    int cd = threadIdx.x, hd = cd >> 5, d = cd & 31;
    const float* wrow = wproj + (size_t)o * C_ + hd * CH_;
    const float* Sblk = S + ((size_t)b * HEADS_ + hd) * CH_ * CH_;
    float acc = 0.0f;
#pragma unroll
    for (int c = 0; c < CH_; ++c) acc += wrow[c] * Sblk[c * CH_ + d];
    Mb[((size_t)b * C_ + o) * C_ + cd] = f2bf(acc);
}

// ---------------- kernel 5: out[b] = M[b] (192x192) @ V[b] (192x16384) ------
// block = 128 threads (4 waves); tile 64(M) x 64(N); K chunks of 32.
// V chunk is transposed into LDS at staging time ([n][k] layout), so B
// fragments are two aligned ds_load_b128 per lane instead of 16 u16 gathers.
__global__ __launch_bounds__(128) void gemm_mv_kernel(
    const unsigned short* __restrict__ Mb,
    const unsigned short* __restrict__ Vb,
    float* __restrict__ out)
{
    constexpr int TLD = 40;                        // 80B row stride: 16B aligned
    __shared__ __align__(16) unsigned short Vt[64 * TLD];   // [n][k], 5120 B

    int blk    = blockIdx.x;
    int nt     = blk & 255;
    int mt     = (blk >> 8) % 3;
    int b      = blk / (256 * 3);
    int n_base = nt * 64, m_base = mt * 64;
    int tid    = threadIdx.x, lane = tid & 31, wave = tid >> 5;

    const unsigned short* Mbase = Mb + ((size_t)b * C_ + m_base + wave * 16) * C_;
    const unsigned short* Vbase = Vb + ((size_t)b * C_) * HW_ + n_base;

    v8f acc[4] = {};

    int kk = tid >> 2;                 // 0..31 : K row within chunk
    int nn = (tid & 3) * 16;           // 0,16,32,48 : N offset within tile
    for (int k0 = 0; k0 < C_; k0 += 32) {
        // coalesced global read of V chunk row, transpose-scatter into LDS
        const unsigned short* src = Vbase + (size_t)(k0 + kk) * HW_ + nn;
        u16x8 lo = *(const u16x8*)(src);
        u16x8 hi = *(const u16x8*)(src + 8);
        if (k0 + 32 < C_)              // global_prefetch next K chunk
            __builtin_prefetch(Vbase + (size_t)(k0 + 32 + kk) * HW_ + nn, 0, 1);
#pragma unroll
        for (int i = 0; i < 8; ++i) {
            Vt[(nn + i)     * TLD + kk] = lo[i];
            Vt[(nn + 8 + i) * TLD + kk] = hi[i];
        }
        __syncthreads();

        v16bf a = fragA(Mbase, C_, 0, k0, lane);   // reused across 4 N tiles
#pragma unroll
        for (int j = 0; j < 4; ++j) {
            v16bf bf = fragBt_lds(Vt, TLD, j * 16, lane);
            acc[j] = wmma_bf16(a, bf, acc[j]);
        }
        __syncthreads();
    }

    int half = lane >> 4, ncol = lane & 15;
    float* obase = out + ((size_t)b * C_ + m_base + wave * 16) * HW_ + n_base;
#pragma unroll
    for (int j = 0; j < 4; ++j)
#pragma unroll
        for (int r = 0; r < 8; ++r)
            obase[(size_t)(r + 8 * half) * HW_ + j * 16 + ncol] = acc[j][r];
}

// ---------------- host-side launcher ----------------
extern "C" void kernel_launch(void* const* d_in, const int* in_sizes, int n_in,
                              void* d_out, int out_size, void* d_ws, size_t ws_size,
                              hipStream_t stream)
{
    const float* x     = (const float*)d_in[0];
    const float* y     = (const float*)d_in[1];
    const float* wdw   = (const float*)d_in[2];
    const float* wproj = (const float*)d_in[3];
    const float* temp  = (const float*)d_in[4];
    float*       out   = (float*)d_out;

    // workspace layout (bf16 q/k/v + bf16 M + small f32 region)
    unsigned short* Qb = (unsigned short*)d_ws;
    unsigned short* Kb = Qb + NQKV;
    unsigned short* Vb = Kb + NQKV;
    unsigned short* Mb = Vb + NQKV;
    float* fbase = (float*)(Mb + NMB);
    float* ssq   = fbase;                 // [B_*C_]
    float* ssk   = fbase + NSSQ;          // [B_*C_]
    float* S     = ssk + NSSQ;            // [B_*HEADS_*CH_*CH_]

    const int NZ = 2 * NSSQ + NS;
    zero_kernel<<<(NZ + 255) / 256, 256, 0, stream>>>(fbase, NZ);

    dwconv_kernel<<<B_ * 3 * C_ * 64, 256, 0, stream>>>(
        x, y, wdw, Qb, Kb, Vb, ssq, ssk);

    attn_qk_kernel<<<B_ * HEADS_ * KSPLIT_, 32, 0, stream>>>(Qb, Kb, S);

    softmax_kernel<<<B_ * HEADS_, 32, 0, stream>>>(S, ssq, ssk, temp);

    buildm_kernel<<<B_ * C_, C_, 0, stream>>>(wproj, S, Mb);

    gemm_mv_kernel<<<B_ * 3 * 256, 128, 0, stream>>>(Mb, Vb, out);
}
